// CAS_Decoder_68719476736200
// MI455X (gfx1250) — matrix operand order repaired and checked by reading, not compiled, not run on hardware
//
#include <hip/hip_runtime.h>
#include <hip/hip_bf16.h>
#include <math.h>

#define Hd 1024
#define Bd 128
#define Ld 96
#define Cd 3
#define Ad 32
#define Sd 64
#define DEGd 6

typedef __attribute__((ext_vector_type(16))) __bf16 v16bf;
typedef __attribute__((ext_vector_type(8)))  __bf16 v8bf;
typedef __attribute__((ext_vector_type(8)))  float  v8f;

struct Ptr6 { const float* p[6]; };
struct CatSrc { const float* p; int len; };
struct Cat8 { CatSrc s[8]; int n; };

__device__ __forceinline__ __bf16 f2bf(float x) { return (__bf16)x; }

__device__ __forceinline__ v16bf cat16(v8bf lo, v8bf hi) {
  return __builtin_shufflevector(lo, hi, 0,1,2,3,4,5,6,7,8,9,10,11,12,13,14,15);
}

// LDS tile row stride in bf16 elements: 40 -> 80 bytes (16B aligned, bank skewed)
#define LDS_STRIDE 40

// A fragment 16x32 bf16 (per ISA layout): lanes 0-15 hold row m0+lane, K {0..7,16..23};
// lanes 16-31 hold row m0+lane-16, K {8..15,24..31}.
__device__ __forceinline__ v16bf load_fragA(const __bf16* tile, int lane) {
  const __bf16* p = tile + (lane & 15) * LDS_STRIDE + ((lane < 16) ? 0 : 8);
  v8bf lo = *(const v8bf*)(p);
  v8bf hi = *(const v8bf*)(p + 16);
  return cat16(lo, hi);
}

// B fragment 32x16 bf16, tile stored as [n][k]: lanes 0-15 -> col n0+lane K {0..7,16..23};
// lanes 16-31 -> col n0+lane-16, K {8..15,24..31}.
__device__ __forceinline__ v16bf load_fragB(const __bf16* tile, int lane) {
  const __bf16* p = tile + (lane & 15) * LDS_STRIDE + ((lane < 16) ? 0 : 8);
  v8bf lo = *(const v8bf*)(p);
  v8bf hi = *(const v8bf*)(p + 16);
  return cat16(lo, hi);
}

__device__ __forceinline__ v8f wmma_bf16(v16bf a, v16bf b, v8f c) {
  return __builtin_amdgcn_wmma_f32_16x16x32_bf16(false, a, false, b, (short)0, c,
                                                 false, false);
}

// ---------------------------------------------------------------------------
// Generic WMMA GEMM: Y[M=128,N] = act(X[128,Kpad(padded)] @ W[N,Kact].T + bias)
// Block tile 128(M) x 64(N); 8 waves arranged 4(M) x 2(N); each wave 32x32
// -> 4 WMMAs per 32-wide K step. X rows zero-padded to Kpad (mult of 32).
// ACT: 0 = none, 1 = tanh.  grid = (ceil(N/64), 1), block = 256.
// ---------------------------------------------------------------------------
template<int ACT>
__global__ __launch_bounds__(256)
void gemm_kernel(const float* __restrict__ X, int ldx,
                 const float* __restrict__ W, int ldw,
                 const float* __restrict__ bias,
                 float* __restrict__ Y,
                 int N, int Kact, int Kpad)
{
  __shared__ __align__(16) __bf16 As[128][LDS_STRIDE];
  __shared__ __align__(16) __bf16 Bs[64][LDS_STRIDE];
  const int tid  = threadIdx.x;
  const int lane = tid & 31, wv = tid >> 5;
  const int m0 = (wv & 3) * 32, n0 = (wv >> 2) * 32;
  const int nBase = blockIdx.x * 64;
  const int lmA = tid >> 1, lkA = (tid & 1) * 16;   // A: 128 rows x 32 K, 16 f32/thread
  const int lmB = tid >> 2, lkB = (tid & 3) * 8;    // B:  64 rows x 32 K,  8 f32/thread

  v8f acc00 = {0.f,0.f,0.f,0.f,0.f,0.f,0.f,0.f};
  v8f acc01 = acc00, acc10 = acc00, acc11 = acc00;

  for (int kt = 0; kt < Kpad; kt += 32) {
    {  // A tile (X zero-padded -> unguarded), rows 0..127 == batch
      const float* src = X + (size_t)lmA * ldx + kt + lkA;
      v8bf t0, t1;
#pragma unroll
      for (int i = 0; i < 8; ++i) { t0[i] = f2bf(src[i]); t1[i] = f2bf(src[8 + i]); }
      *(v8bf*)&As[lmA][lkA]     = t0;
      *(v8bf*)&As[lmA][lkA + 8] = t1;
    }
    {  // W tile (guarded)
      const int gn = nBase + lmB;
      v8bf t;
#pragma unroll
      for (int i = 0; i < 8; ++i) {
        const int gk = kt + lkB + i;
        float w = 0.f;
        if (gn < N && gk < Kact) w = W[(size_t)gn * ldw + gk];
        t[i] = f2bf(w);
      }
      *(v8bf*)&Bs[lmB][lkB] = t;
    }
    __syncthreads();
    v16bf a0 = load_fragA(&As[m0][0], lane);
    v16bf a1 = load_fragA(&As[m0 + 16][0], lane);
    v16bf b0 = load_fragB(&Bs[n0][0], lane);
    v16bf b1 = load_fragB(&Bs[n0 + 16][0], lane);
    acc00 = wmma_bf16(a0, b0, acc00);
    acc01 = wmma_bf16(a0, b1, acc01);
    acc10 = wmma_bf16(a1, b0, acc10);
    acc11 = wmma_bf16(a1, b1, acc11);
    __syncthreads();
  }

  // C layout: VGPR i, lanes 0-15 -> M=sub_m+i, lanes 16-31 -> M=sub_m+8+i; N = lane&15
#pragma unroll
  for (int ai = 0; ai < 2; ++ai) {
    const int mrow = m0 + ai * 16 + ((lane >= 16) ? 8 : 0);
#pragma unroll
    for (int bi = 0; bi < 2; ++bi) {
      const v8f acc = ai ? (bi ? acc11 : acc10) : (bi ? acc01 : acc00);
      const int n = nBase + n0 + bi * 16 + (lane & 15);
      if (n < N) {
        const float bv = bias ? bias[n] : 0.f;
#pragma unroll
        for (int i = 0; i < 8; ++i) {
          float val = acc[i] + bv;
          if (ACT == 1) val = tanhf(val);
          Y[(size_t)(mrow + i) * N + n] = val;
        }
      }
    }
  }
}

// ---------------------------------------------------------------------------
// Fused attention-score GEMM: for stream j (blockIdx.z):
//   penc = enc_j[r=l*B+b, :] @ We_j.T   (We_j = attn_W[stage,j][:, H:2H])
//   score[j][r] += sum_n v[n] * tanh(penc[r,n] + qp[j][b,n])   (qp includes bias)
// Block tile 128(M) x 64(N), 4 WMMAs per wave per K step.
// grid = (H/64, L*B/128, 6), block = 256
// ---------------------------------------------------------------------------
__global__ __launch_bounds__(256)
void attn_scores_kernel(Ptr6 encs,
                        const float* __restrict__ attn_W, int stage,
                        const float* __restrict__ qp,
                        const float* __restrict__ attn_v,
                        float* __restrict__ scores)
{
  __shared__ __align__(16) __bf16 As[128][LDS_STRIDE];
  __shared__ __align__(16) __bf16 Bs[64][LDS_STRIDE];
  const int j = blockIdx.z;
  const float* __restrict__ E  = encs.p[j];
  const float* __restrict__ Wj = attn_W + ((size_t)(stage * 6 + j) * Hd) * (2 * Hd) + Hd;
  const float* __restrict__ QP = qp + (size_t)j * Bd * Hd;
  const float* __restrict__ V  = attn_v + (size_t)(stage * 6 + j) * Hd;
  float* __restrict__ SC = scores + (size_t)j * (Ld * Bd);

  const int tid  = threadIdx.x;
  const int lane = tid & 31, wv = tid >> 5;
  const int m0 = (wv & 3) * 32, n0 = (wv >> 2) * 32;
  const int mBase = blockIdx.y * 128, nBase = blockIdx.x * 64;
  const int lmA = tid >> 1, lkA = (tid & 1) * 16;
  const int lmB = tid >> 2, lkB = (tid & 3) * 8;

  v8f acc00 = {0.f,0.f,0.f,0.f,0.f,0.f,0.f,0.f};
  v8f acc01 = acc00, acc10 = acc00, acc11 = acc00;

  const float* srcA = E  + (size_t)(mBase + lmA) * Hd + lkA;
  const float* srcB = Wj + (size_t)(nBase + lmB) * (2 * Hd) + lkB;

  for (int kt = 0; kt < Hd; kt += 32) {
    float4 a0 = *(const float4*)(srcA + kt);
    float4 a1 = *(const float4*)(srcA + kt + 4);
    float4 a2 = *(const float4*)(srcA + kt + 8);
    float4 a3 = *(const float4*)(srcA + kt + 12);
    float4 b0 = *(const float4*)(srcB + kt);
    float4 b1 = *(const float4*)(srcB + kt + 4);
    if (kt + 32 < Hd) {  // emits global_prefetch_b8 on gfx1250
      __builtin_prefetch(srcA + kt + 32, 0, 1);
      __builtin_prefetch(srcB + kt + 32, 0, 1);
    }
    v8bf ta0, ta1, tb;
    ta0[0] = f2bf(a0.x); ta0[1] = f2bf(a0.y); ta0[2] = f2bf(a0.z); ta0[3] = f2bf(a0.w);
    ta0[4] = f2bf(a1.x); ta0[5] = f2bf(a1.y); ta0[6] = f2bf(a1.z); ta0[7] = f2bf(a1.w);
    ta1[0] = f2bf(a2.x); ta1[1] = f2bf(a2.y); ta1[2] = f2bf(a2.z); ta1[3] = f2bf(a2.w);
    ta1[4] = f2bf(a3.x); ta1[5] = f2bf(a3.y); ta1[6] = f2bf(a3.z); ta1[7] = f2bf(a3.w);
    tb[0]  = f2bf(b0.x); tb[1]  = f2bf(b0.y); tb[2]  = f2bf(b0.z); tb[3]  = f2bf(b0.w);
    tb[4]  = f2bf(b1.x); tb[5]  = f2bf(b1.y); tb[6]  = f2bf(b1.z); tb[7]  = f2bf(b1.w);
    *(v8bf*)&As[lmA][lkA]     = ta0;
    *(v8bf*)&As[lmA][lkA + 8] = ta1;
    *(v8bf*)&Bs[lmB][lkB]     = tb;
    __syncthreads();
    v16bf fa0 = load_fragA(&As[m0][0], lane);
    v16bf fa1 = load_fragA(&As[m0 + 16][0], lane);
    v16bf fb0 = load_fragB(&Bs[n0][0], lane);
    v16bf fb1 = load_fragB(&Bs[n0 + 16][0], lane);
    acc00 = wmma_bf16(fa0, fb0, acc00);
    acc01 = wmma_bf16(fa0, fb1, acc01);
    acc10 = wmma_bf16(fa1, fb0, acc10);
    acc11 = wmma_bf16(fa1, fb1, acc11);
    __syncthreads();
  }

  // Epilogue: tanh + dot with v + per-row reduction over this tile's 16 columns
#pragma unroll
  for (int ai = 0; ai < 2; ++ai) {
    const int mrow = mBase + m0 + ai * 16 + ((lane >= 16) ? 8 : 0);
#pragma unroll
    for (int bi = 0; bi < 2; ++bi) {
      const v8f acc = ai ? (bi ? acc11 : acc10) : (bi ? acc01 : acc00);
      const int n = nBase + n0 + bi * 16 + (lane & 15);
      const float vn = V[n];
#pragma unroll
      for (int i = 0; i < 8; ++i) {
        const int r = mrow + i;           // r = l*B + b
        const int b = r & (Bd - 1);
        float t  = tanhf(acc[i] + QP[(size_t)b * Hd + n]);
        float pv = vn * t;
#pragma unroll
        for (int off = 1; off < 16; off <<= 1) pv += __shfl_xor(pv, off, 32);
        if ((lane & 15) == 0) atomicAdd(&SC[r], pv);
      }
    }
  }
}

// Softmax over L=96 per (stream, b). One wave per row, 3 elements per lane.
__global__ void attn_softmax_kernel(const float* __restrict__ scores,
                                    float* __restrict__ alpha)
{
  const int row = blockIdx.x;               // j*B + b
  const int j = row >> 7, b = row & (Bd - 1);
  const float* SC = scores + (size_t)j * Ld * Bd;
  const int lane = threadIdx.x;
  float s0 = SC[(size_t)(lane)      * Bd + b];
  float s1 = SC[(size_t)(lane + 32) * Bd + b];
  float s2 = SC[(size_t)(lane + 64) * Bd + b];
  float mx = fmaxf(s0, fmaxf(s1, s2));
#pragma unroll
  for (int off = 16; off > 0; off >>= 1) mx = fmaxf(mx, __shfl_xor(mx, off, 32));
  float e0 = __expf(s0 - mx), e1 = __expf(s1 - mx), e2 = __expf(s2 - mx);
  float sm = e0 + e1 + e2;
#pragma unroll
  for (int off = 16; off > 0; off >>= 1) sm += __shfl_xor(sm, off, 32);
  const float inv = 1.f / sm;
  float* AL = alpha + (size_t)row * Ld;
  AL[lane] = e0 * inv; AL[lane + 32] = e1 * inv; AL[lane + 64] = e2 * inv;
}

// context[j][b][h] = sum_l alpha[j,b,l] * enc_j[l,b,h]
__global__ void attn_context_kernel(Ptr6 encs, const float* __restrict__ alpha,
                                    float* __restrict__ ctx)
{
  const int j = blockIdx.z, b = blockIdx.y;
  const int hh = blockIdx.x * 256 + threadIdx.x;
  __shared__ float a[Ld];
  if (threadIdx.x < Ld) a[threadIdx.x] = alpha[((size_t)j * Bd + b) * Ld + threadIdx.x];
  __syncthreads();
  const float* E = encs.p[j] + (size_t)b * Hd + hh;
  float acc = 0.f;
#pragma unroll 4
  for (int l = 0; l < Ld; ++l) acc += a[l] * E[(size_t)l * Bd * Hd];
  ctx[((size_t)j * Bd + b) * Hd + hh] = acc;
}

// Concatenate up to 8 [B, len] segments into [B, Kpad] row-major, zero padded.
__global__ void concat_kernel(Cat8 cat, float* __restrict__ dst, int Kpad)
{
  const int b = blockIdx.y;
  const int k = blockIdx.x * 256 + threadIdx.x;
  if (k >= Kpad) return;
  float v = 0.f; int off = 0;
#pragma unroll
  for (int s = 0; s < 8; ++s) {
    if (s < cat.n) {
      const int len = cat.s[s].len;
      if (k >= off && k < off + len) v = cat.s[s].p[(size_t)b * len + (k - off)];
      off += len;
    }
  }
  dst[(size_t)b * Kpad + k] = v;
}

// GRU gate combine: h = (1-z)*n + z*h ; gi/gh already include bih/bhh
__global__ void gru_gate_kernel(const float* __restrict__ gi,
                                const float* __restrict__ gh,
                                float* __restrict__ h)
{
  const int idx = blockIdx.x * 256 + threadIdx.x;   // b*H + hh
  const int b = idx >> 10, hh = idx & (Hd - 1);
  const float* GI = gi + (size_t)b * 3 * Hd;
  const float* GH = gh + (size_t)b * 3 * Hd;
  const float r = 1.f / (1.f + __expf(-(GI[hh] + GH[hh])));
  const float z = 1.f / (1.f + __expf(-(GI[Hd + hh] + GH[Hd + hh])));
  const float n = tanhf(GI[2 * Hd + hh] + r * GH[2 * Hd + hh]);
  h[idx] = (1.f - z) * n + z * h[idx];
}

__global__ void softmax_rows_kernel(const float* __restrict__ Yin, int N,
                                    float* __restrict__ o1, float* __restrict__ o2)
{
  const int b = blockIdx.x, lane = threadIdx.x;
  float v = (lane < N) ? Yin[b * N + lane] : -3.0e38f;
  float mx = v;
#pragma unroll
  for (int off = 16; off > 0; off >>= 1) mx = fmaxf(mx, __shfl_xor(mx, off, 32));
  float e = (lane < N) ? __expf(v - mx) : 0.f;
  float sm = e;
#pragma unroll
  for (int off = 16; off > 0; off >>= 1) sm += __shfl_xor(sm, off, 32);
  if (lane < N) {
    const float r = e / sm;
    o1[b * N + lane] = r;
    o2[b * N + lane] = r;
  }
}

__global__ void sigmoid_kernel(const float* __restrict__ Yin, int n,
                               float* __restrict__ out)
{
  const int i = blockIdx.x * 256 + threadIdx.x;
  if (i < n) out[i] = 1.f / (1.f + __expf(-Yin[i]));
}

__global__ void copy_kernel(const float* __restrict__ src, float* __restrict__ dst, int n)
{
  const int i = blockIdx.x * 256 + threadIdx.x;
  if (i < n) dst[i] = src[i];
}

__global__ void zero_kernel(float* __restrict__ dst, int n)
{
  const int i = blockIdx.x * 256 + threadIdx.x;
  if (i < n) dst[i] = 0.f;
}

// ---------------------------------------------------------------------------
extern "C" void kernel_launch(void* const* d_in, const int* in_sizes, int n_in,
                              void* d_out, int out_size, void* d_ws, size_t ws_size,
                              hipStream_t stream)
{
  (void)in_sizes; (void)n_in; (void)out_size; (void)ws_size;

  // Reference stream order: a, user_inform, user_request, agent_propose, agent_request, u
  Ptr6 encs;
  encs.p[0] = (const float*)d_in[0];
  encs.p[1] = (const float*)d_in[2];
  encs.p[2] = (const float*)d_in[1];
  encs.p[3] = (const float*)d_in[4];
  encs.p[4] = (const float*)d_in[3];
  encs.p[5] = (const float*)d_in[5];
  const float* q    = (const float*)d_in[6];
  const float* kb   = (const float*)d_in[7];
  const float* lc   = (const float*)d_in[8];
  const float* la   = (const float*)d_in[9];
  const float* ls   = (const float*)d_in[10];
  const float* h0   = (const float*)d_in[11];
  const float* cont_in_W  = (const float*)d_in[12];
  const float* cont_in_b  = (const float*)d_in[13];
  const float* act_in_W   = (const float*)d_in[14];
  const float* act_in_b   = (const float*)d_in[15];
  const float* slot_in_W  = (const float*)d_in[16];
  const float* slot_in_b  = (const float*)d_in[17];
  const float* cont_out_W = (const float*)d_in[18];
  const float* cont_out_b = (const float*)d_in[19];
  const float* act_out_W  = (const float*)d_in[20];
  const float* act_out_b  = (const float*)d_in[21];
  const float* slot_out_W = (const float*)d_in[22];
  const float* slot_out_b = (const float*)d_in[23];
  const float* attn_W = (const float*)d_in[24];
  const float* attn_b = (const float*)d_in[25];
  const float* attn_v = (const float*)d_in[26];
  const float* gru_Wih = (const float*)d_in[27];
  const float* gru_Whh = (const float*)d_in[28];
  const float* gru_bih = (const float*)d_in[29];
  const float* gru_bhh = (const float*)d_in[30];

  float* ws = (float*)d_ws;
  size_t o = 0;
  float* ws_scores = ws + o; o += (size_t)6 * Ld * Bd;
  float* ws_alpha  = ws + o; o += (size_t)6 * Bd * Ld;
  float* ws_qp     = ws + o; o += (size_t)6 * Bd * Hd;
  float* ws_ctx    = ws + o; o += (size_t)6 * Bd * Hd;
  float* ws_x      = ws + o; o += (size_t)Bd * Hd;
  float* ws_xcat   = ws + o; o += (size_t)Bd * 1152;
  float* ws_gin    = ws + o; o += (size_t)Bd * 7200;
  float* ws_gi     = ws + o; o += (size_t)Bd * 3 * Hd;
  float* ws_gh     = ws + o; o += (size_t)Bd * 3 * Hd;
  float* ws_ocat   = ws + o; o += (size_t)Bd * 7168;
  float* ws_y      = ws + o; o += (size_t)Bd * 64;
  float* ws_cont   = ws + o; o += (size_t)Bd * Cd;
  float* ws_act    = ws + o; o += (size_t)Bd * Ad;
  float* ws_h      = ws + o; o += (size_t)Bd * Hd;   // ~19 MB total

  float* out_cont = (float*)d_out;
  float* out_act  = out_cont + Bd * Cd;
  float* out_slot = out_act + Bd * Ad;
  float* out_h    = out_slot + Bd * Sd;

  copy_kernel<<<(Bd * Hd) / 256, 256, 0, stream>>>(h0, ws_h, Bd * Hd);

  const int perm0[6] = {0, 1, 2, 4, 3, 5};   // stage-0 context cat order
  const int permI[6] = {0, 1, 2, 3, 4, 5};

  for (int stage = 0; stage < 3; ++stage) {
    // ---- stage input projection: x = tanh(cat @ inW.T + inB) ----
    int Kact, Kpad;
    const float *inW, *inB;
    Cat8 cat;
    for (int s = 0; s < 8; ++s) { cat.s[s].p = nullptr; cat.s[s].len = 0; }
    if (stage == 0) {
      cat.n = 4;
      cat.s[0].p = q;       cat.s[0].len = Hd;
      cat.s[1].p = lc;      cat.s[1].len = Cd;
      cat.s[2].p = la;      cat.s[2].len = Ad;
      cat.s[3].p = ls;      cat.s[3].len = Sd;
      Kact = Hd + Cd + Ad + Sd; Kpad = 1152; inW = cont_in_W; inB = cont_in_b;
    } else if (stage == 1) {
      cat.n = 3;
      cat.s[0].p = ws_cont; cat.s[0].len = Cd;
      cat.s[1].p = la;      cat.s[1].len = Ad;
      cat.s[2].p = ls;      cat.s[2].len = Sd;
      Kact = Cd + Ad + Sd; Kpad = 128; inW = act_in_W; inB = act_in_b;
    } else {
      cat.n = 4;
      cat.s[0].p = ws_cont; cat.s[0].len = Cd;
      cat.s[1].p = ws_act;  cat.s[1].len = Ad;
      cat.s[2].p = la;      cat.s[2].len = Ad;
      cat.s[3].p = ls;      cat.s[3].len = Sd;
      Kact = Cd + 2 * Ad + Sd; Kpad = 160; inW = slot_in_W; inB = slot_in_b;
    }
    concat_kernel<<<dim3((Kpad + 255) / 256, Bd), 256, 0, stream>>>(cat, ws_xcat, Kpad);
    gemm_kernel<1><<<dim3(Hd / 64, 1), 256, 0, stream>>>(
        ws_xcat, Kpad, inW, Kact, inB, ws_x, Hd, Kact, Kpad);

    // ---- per-stream query projections qp = x @ Wq.T + attn_b ----
    for (int j = 0; j < 6; ++j) {
      const float* Wq = attn_W + (size_t)(stage * 6 + j) * Hd * 2 * Hd;
      const float* bq = attn_b + (size_t)(stage * 6 + j) * Hd;
      gemm_kernel<0><<<dim3(Hd / 64, 1), 256, 0, stream>>>(
          ws_x, Hd, Wq, 2 * Hd, bq, ws_qp + (size_t)j * Bd * Hd, Hd, Hd, Hd);
    }

    // ---- fused attention: enc @ We.T -> tanh -> dot v -> scores ----
    zero_kernel<<<(6 * Ld * Bd) / 256, 256, 0, stream>>>(ws_scores, 6 * Ld * Bd);
    attn_scores_kernel<<<dim3(Hd / 64, (Ld * Bd) / 128, 6), 256, 0, stream>>>(
        encs, attn_W, stage, ws_qp, attn_v, ws_scores);
    attn_softmax_kernel<<<6 * Bd, 32, 0, stream>>>(ws_scores, ws_alpha);
    attn_context_kernel<<<dim3(Hd / 256, Bd, 6), 256, 0, stream>>>(encs, ws_alpha, ws_ctx);

    // ---- GRU step ----
    const int* pm = (stage == 0) ? perm0 : permI;
    Cat8 g; g.n = 8;
    g.s[0].p = kb;   g.s[0].len = DEGd;
    g.s[1].p = ws_x; g.s[1].len = Hd;
    for (int t = 0; t < 6; ++t) { g.s[2 + t].p = ws_ctx + (size_t)pm[t] * Bd * Hd; g.s[2 + t].len = Hd; }
    concat_kernel<<<dim3((7200 + 255) / 256, Bd), 256, 0, stream>>>(g, ws_gin, 7200);
    const int Kg = 7 * Hd + DEGd;  // 7174
    gemm_kernel<0><<<dim3((3 * Hd) / 64, 1), 256, 0, stream>>>(
        ws_gin, 7200, gru_Wih + (size_t)stage * 3 * Hd * Kg, Kg,
        gru_bih + (size_t)stage * 3 * Hd, ws_gi, 3 * Hd, Kg, 7200);
    gemm_kernel<0><<<dim3((3 * Hd) / 64, 1), 256, 0, stream>>>(
        ws_h, Hd, gru_Whh + (size_t)stage * 3 * Hd * Hd, Hd,
        gru_bhh + (size_t)stage * 3 * Hd, ws_gh, 3 * Hd, Hd, Hd);
    gru_gate_kernel<<<(Bd * Hd) / 256, 256, 0, stream>>>(ws_gi, ws_gh, ws_h);

    // ---- output head ----
    Cat8 oc; oc.n = 7;
    oc.s[0].p = ws_h; oc.s[0].len = Hd;
    for (int t = 0; t < 6; ++t) { oc.s[1 + t].p = ws_ctx + (size_t)pm[t] * Bd * Hd; oc.s[1 + t].len = Hd; }
    oc.s[7].p = nullptr; oc.s[7].len = 0;
    concat_kernel<<<dim3(7168 / 256, Bd), 256, 0, stream>>>(oc, ws_ocat, 7168);
    if (stage == 0) {
      gemm_kernel<0><<<dim3(1, 1), 256, 0, stream>>>(
          ws_ocat, 7168, cont_out_W, 7168, cont_out_b, ws_y, Cd, 7168, 7168);
      softmax_rows_kernel<<<Bd, 32, 0, stream>>>(ws_y, Cd, ws_cont, out_cont);
    } else if (stage == 1) {
      gemm_kernel<0><<<dim3(1, 1), 256, 0, stream>>>(
          ws_ocat, 7168, act_out_W, 7168, act_out_b, ws_y, Ad, 7168, 7168);
      softmax_rows_kernel<<<Bd, 32, 0, stream>>>(ws_y, Ad, ws_act, out_act);
    } else {
      gemm_kernel<0><<<dim3(1, 1), 256, 0, stream>>>(
          ws_ocat, 7168, slot_out_W, 7168, slot_out_b, ws_y, Sd, 7168, 7168);
      sigmoid_kernel<<<(Bd * Sd) / 256, 256, 0, stream>>>(ws_y, Bd * Sd, out_slot);
    }
  }

  copy_kernel<<<(Bd * Hd) / 256, 256, 0, stream>>>(ws_h, out_h, Bd * Hd);
}
